// DiTBlock_21500606283921
// MI455X (gfx1250) — matrix operand order repaired and checked by reading, not compiled, not run on hardware
//
#include <hip/hip_runtime.h>
#include <stdint.h>

typedef unsigned short ush;
typedef __attribute__((ext_vector_type(16))) __bf16 v16bf;
typedef __attribute__((ext_vector_type(8)))  float  v8f;

#define D      1152
#define S      1024
#define BB     8
#define NH     16
#define HD     72
#define HID    4608
#define M_ROWS (BB*S)
#define SIXD   (6*D)

#define EP_BF16      0
#define EP_GELU_BF16 1
#define EP_F32       2

#if __has_builtin(__builtin_amdgcn_global_load_async_to_lds_b128) && \
    __has_builtin(__builtin_amdgcn_s_wait_asynccnt)
#define HAVE_ASYNC 1
#define AS1 __attribute__((address_space(1)))
#define AS3 __attribute__((address_space(3)))
typedef int v4i_ __attribute__((vector_size(16)));
#define ASYNC_CP16(g, l) \
  __builtin_amdgcn_global_load_async_to_lds_b128((AS1 v4i_*)(g), (AS3 v4i_*)(l), 0, 0)
#endif

__device__ __forceinline__ ush f2bf(float f) {
  unsigned u = __float_as_uint(f);
  return (ush)((u + 0x7FFFu + ((u >> 16) & 1u)) >> 16);
}

struct Frag { union { v16bf v; uint4 u[2]; }; };

// ---------------------------------------------------------------- elementwise
// fused transpose + fp32->bf16: out[n*K + k] = bf16(in[k*N + n])
__global__ __launch_bounds__(256) void k_w_t_bf16(const float* __restrict__ in,
                                                  ush* __restrict__ out, int K, int N) {
  __shared__ float t[32][33];
  int n0 = blockIdx.x * 32, k0 = blockIdx.y * 32;
  int tx = threadIdx.x & 31, ty = threadIdx.x >> 5;   // 32 x 8
  for (int i = 0; i < 32; i += 8)
    t[ty + i][tx] = in[(size_t)(k0 + ty + i) * N + n0 + tx];
  __syncthreads();
  for (int i = 0; i < 32; i += 8)
    out[(size_t)(n0 + ty + i) * K + k0 + tx] = f2bf(t[tx][ty + i]);
}

// mod[b, j] = silu(c[b,:]) @ ada_w[:, j] + ada_b[j]
__global__ __launch_bounds__(256) void k_ada(const float* __restrict__ c,
                                             const float* __restrict__ w,
                                             const float* __restrict__ bias,
                                             float* __restrict__ mod) {
  int j = blockIdx.x * 256 + threadIdx.x;
  int b = blockIdx.y;
  if (j >= SIXD) return;
  float acc = bias[j];
  for (int k = 0; k < D; ++k) {
    float cv = c[b * D + k];
    float sl = cv / (1.f + __expf(-cv));
    acc = fmaf(sl, w[(size_t)k * SIXD + j], acc);
  }
  mod[b * SIXD + j] = acc;
}

// h = modulate(layer_norm(x)) -> bf16
__global__ __launch_bounds__(256) void k_ln_mod(const float* __restrict__ x,
                                                const float* __restrict__ w,
                                                const float* __restrict__ bias,
                                                const float* __restrict__ mod,
                                                int shOff, int scOff,
                                                ush* __restrict__ out) {
  __shared__ float red[8];
  int row = blockIdx.x;
  int bidx = row / S;
  int tid = threadIdx.x, lane = tid & 31, wid = tid >> 5;
  const float* xr = x + (size_t)row * D;

  float xc[5];
  int cnt = 0;
  float s0 = 0.f;
  for (int cI = tid; cI < D; cI += 256) { float v = xr[cI]; xc[cnt++] = v; s0 += v; }
  for (int o = 16; o > 0; o >>= 1) s0 += __shfl_xor(s0, o, 32);
  if (lane == 0) red[wid] = s0;
  __syncthreads();
  if (tid == 0) { float t = 0.f; for (int i = 0; i < 8; ++i) t += red[i]; red[0] = t; }
  __syncthreads();
  float mean = red[0] * (1.f / D);
  __syncthreads();

  float s1 = 0.f;
  for (int k = 0; k < cnt; ++k) { float d = xc[k] - mean; s1 += d * d; }
  for (int o = 16; o > 0; o >>= 1) s1 += __shfl_xor(s1, o, 32);
  if (lane == 0) red[wid] = s1;
  __syncthreads();
  if (tid == 0) { float t = 0.f; for (int i = 0; i < 8; ++i) t += red[i]; red[0] = t; }
  __syncthreads();
  float rstd = rsqrtf(red[0] * (1.f / D) + 1e-5f);

  const float* shp = mod + (size_t)bidx * SIXD + shOff;
  const float* scp = mod + (size_t)bidx * SIXD + scOff;
  int k = 0;
  for (int cI = tid; cI < D; cI += 256, ++k) {
    float nv = (xc[k] - mean) * rstd;
    float mv = nv * w[cI] + bias[cI];
    out[(size_t)row * D + cI] = f2bf(mv * (1.f + scp[cI]) + shp[cI]);
  }
}

// out = x + g * p
__global__ void k_residual(const float* __restrict__ x, const float* __restrict__ p,
                           const float* __restrict__ mod, int gOff,
                           float* __restrict__ out) {
  size_t i = (size_t)blockIdx.x * 256 + threadIdx.x;
  if (i >= (size_t)M_ROWS * D) return;
  int col = (int)(i % D);
  int bidx = (int)(i / ((size_t)S * D));
  out[i] = x[i] + mod[(size_t)bidx * SIXD + gOff + col] * p[i];
}

// ---------------------------------------------------------------- bf16 WMMA GEMM
// C[M,N] = A[M,K] @ Bt^T (+bias); Bt is pre-transposed [N][K].
// Tiles 128x128x32, ping-pong LDS with branch-free async prefetch.
__global__ __launch_bounds__(256) void k_gemm(const ush* __restrict__ A,
                                              const ush* __restrict__ Bt,
                                              const float* __restrict__ bias,
                                              void* __restrict__ Cout,
                                              int Ndim, int Kdim, int ep) {
  __shared__ ush As[2][128][48];
  __shared__ ush Bs[2][128][48];
  int tid = threadIdx.x, lane = tid & 31, wid = tid >> 5;
  int wm = wid >> 2, wn = wid & 3;
  int r0 = blockIdx.y * 128, c0 = blockIdx.x * 128;

  v8f acc[4][2];
  for (int i = 0; i < 4; ++i)
    for (int j = 0; j < 2; ++j)
      for (int r = 0; r < 8; ++r) acc[i][j][r] = 0.f;

  int srow = tid >> 1, scol = (tid & 1) * 16;
  const ush* gA = A  + (size_t)(r0 + srow) * Kdim + scol;
  const ush* gB = Bt + (size_t)(c0 + srow) * Kdim + scol;

#ifdef HAVE_ASYNC
  ASYNC_CP16(gA,     &As[0][srow][scol]);
  ASYNC_CP16(gA + 8, &As[0][srow][scol + 8]);
  ASYNC_CP16(gB,     &Bs[0][srow][scol]);
  ASYNC_CP16(gB + 8, &Bs[0][srow][scol + 8]);
#else
  uint4 ra0 = *(const uint4*)gA, ra1 = *(const uint4*)(gA + 8);
  uint4 rb0 = *(const uint4*)gB, rb1 = *(const uint4*)(gB + 8);
#endif

  int mr = wm * 64 + (lane & 15);
  int abase = (lane < 16) ? 0 : 8;     // A frag: K{0..7,16..23} / K{8..15,24..31}
  int nb = wn * 32 + (lane & 15);
  int kbase = (lane < 16) ? 0 : 8;     // B frag: K{0..7,16..23} / K{8..15,24..31}

  int buf = 0;
  for (int k0 = 0; k0 < Kdim; k0 += 32) {
    // branch-free prefetch offset: wrap to 0 on the last tile (write is unread)
    int kn = k0 + 32;
    kn = (kn < Kdim) ? kn : 0;
#ifdef HAVE_ASYNC
    __builtin_amdgcn_s_wait_asynccnt(0);
    __syncthreads();
    {
      const ush* nA = gA + kn;
      const ush* nB = gB + kn;
      ASYNC_CP16(nA,     &As[buf ^ 1][srow][scol]);
      ASYNC_CP16(nA + 8, &As[buf ^ 1][srow][scol + 8]);
      ASYNC_CP16(nB,     &Bs[buf ^ 1][srow][scol]);
      ASYNC_CP16(nB + 8, &Bs[buf ^ 1][srow][scol + 8]);
    }
#else
    *(uint4*)&As[buf][srow][scol]     = ra0;
    *(uint4*)&As[buf][srow][scol + 8] = ra1;
    *(uint4*)&Bs[buf][srow][scol]     = rb0;
    *(uint4*)&Bs[buf][srow][scol + 8] = rb1;
    __syncthreads();
    {
      const ush* nA = gA + kn;
      const ush* nB = gB + kn;
      ra0 = *(const uint4*)nA; ra1 = *(const uint4*)(nA + 8);
      rb0 = *(const uint4*)nB; rb1 = *(const uint4*)(nB + 8);
    }
#endif
    Frag af[4], bfr[2];
    for (int i = 0; i < 4; ++i) {
      af[i].u[0] = *(const uint4*)&As[buf][mr + i * 16][abase];
      af[i].u[1] = *(const uint4*)&As[buf][mr + i * 16][abase + 16];
    }
    for (int j = 0; j < 2; ++j) {
      bfr[j].u[0] = *(const uint4*)&Bs[buf][nb + j * 16][kbase];
      bfr[j].u[1] = *(const uint4*)&Bs[buf][nb + j * 16][kbase + 16];
    }
    for (int i = 0; i < 4; ++i)
      for (int j = 0; j < 2; ++j)
        acc[i][j] = __builtin_amdgcn_wmma_f32_16x16x32_bf16(
            false, af[i].v, false, bfr[j].v, (short)0, acc[i][j], false, false);
    buf ^= 1;
  }

  for (int i = 0; i < 4; ++i)
    for (int j = 0; j < 2; ++j) {
      int col = c0 + wn * 32 + j * 16 + (lane & 15);
      float bv = bias ? bias[col] : 0.f;
      for (int r = 0; r < 8; ++r) {
        int row = r0 + wm * 64 + i * 16 + r + ((lane >= 16) ? 8 : 0);
        float v = acc[i][j][r] + bv;
        if (ep == EP_GELU_BF16) v = 0.5f * v * (1.f + erff(v * 0.70710678118f));
        if (ep == EP_F32) ((float*)Cout)[(size_t)row * Ndim + col] = v;
        else             ((ush*)Cout)[(size_t)row * Ndim + col] = f2bf(v);
      }
    }
}

// ---------------------------------------------------------------- flash attention
__global__ __launch_bounds__(256) void k_attn(const ush* __restrict__ Q,
                                              const ush* __restrict__ K,
                                              const ush* __restrict__ V,
                                              ush* __restrict__ O) {
  extern __shared__ ush sm[];
  ush* Ks = sm;                    // [128][104]  key-major, hd contiguous
  ush* Vt = sm + 128 * 104;        // [80][136]   hd-major,  key contiguous
  ush* Ps = Vt + 80 * 136;         // [128][136]  P staging

  int tid = threadIdx.x, lane = tid & 31, wid = tid >> 5;
  int bid = blockIdx.x;
  int qt = bid & 7;
  int h  = (bid >> 3) & 15;
  int b  = bid >> 7;
  int hc = h * HD;
  size_t qrow0 = (size_t)(b * S + qt * 128 + wid * 16);

#ifdef HAVE_ASYNC
  // zero the K pad columns (72..95) once; async copies only write valid data
  for (int i = tid; i < 128 * 24; i += 256)
    Ks[(i / 24) * 104 + 72 + (i % 24)] = 0;
#endif

  Frag aq[3];
  {
    size_t rr = qrow0 + (lane & 15);
    uint4 z; z.x = z.y = z.z = z.w = 0u;
    for (int ks = 0; ks < 3; ++ks) {
      int kb = ks * 32 + ((lane < 16) ? 0 : 8);
      aq[ks].u[0] = (kb + 8 <= HD) ? *(const uint4*)(Q + rr * D + hc + kb) : z;
      int kb2 = kb + 16;
      aq[ks].u[1] = (kb2 + 8 <= HD) ? *(const uint4*)(Q + rr * D + hc + kb2) : z;
    }
  }

  float mrow[8], lrow[8];
  v8f oacc[5];
  for (int r = 0; r < 8; ++r) { mrow[r] = -1e30f; lrow[r] = 0.f; }
  for (int j = 0; j < 5; ++j)
    for (int r = 0; r < 8; ++r) oacc[j][r] = 0.f;
  const float scl = 0.11785113f;   // 1/sqrt(72)

  int nbase = (lane & 15);
  int kqb = (lane < 16) ? 0 : 8;

  for (int kt = 0; kt < 8; ++kt) {
    __syncthreads();
#ifdef HAVE_ASYNC
    for (int i = tid; i < 128 * 9; i += 256) {       // 72 halves = 9 x 16B per key
      int key = i / 9, ch = (i % 9) * 8;
      ASYNC_CP16(K + (size_t)(b * S + kt * 128 + key) * D + hc + ch,
                 &Ks[key * 104 + ch]);
    }
#else
    for (int i = tid; i < 128 * 48; i += 256) {
      int key = i / 48, hdp = (i % 48) * 2;
      unsigned val = 0;
      if (hdp < HD) val = *(const unsigned*)(K + (size_t)(b * S + kt * 128 + key) * D + hc + hdp);
      *(unsigned*)&Ks[key * 104 + hdp] = val;
    }
#endif
    for (int i = tid; i < 128 * 40; i += 256) {
      int key = i / 40, hdp = (i % 40) * 2;
      unsigned val = 0;
      if (hdp < HD) val = *(const unsigned*)(V + (size_t)(b * S + kt * 128 + key) * D + hc + hdp);
      ush* pp = (ush*)&val;
      Vt[hdp * 136 + key] = pp[0];
      Vt[(hdp + 1) * 136 + key] = pp[1];
    }
#ifdef HAVE_ASYNC
    __builtin_amdgcn_s_wait_asynccnt(0);
#endif
    __syncthreads();

    // S = Q @ K^T (16x128 per wave, K=96 in 3 steps)
    v8f sacc[8];
    for (int nt = 0; nt < 8; ++nt)
      for (int r = 0; r < 8; ++r) sacc[nt][r] = 0.f;
    for (int ks = 0; ks < 3; ++ks)
      for (int nt = 0; nt < 8; ++nt) {
        Frag bfr;
        const ush* p = &Ks[(nt * 16 + nbase) * 104 + ks * 32 + kqb];
        bfr.u[0] = *(const uint4*)p;
        bfr.u[1] = *(const uint4*)(p + 16);
        sacc[nt] = __builtin_amdgcn_wmma_f32_16x16x32_bf16(
            false, aq[ks].v, false, bfr.v, (short)0, sacc[nt], false, false);
      }

    // online softmax
    for (int nt = 0; nt < 8; ++nt)
      for (int r = 0; r < 8; ++r) sacc[nt][r] *= scl;
    for (int r = 0; r < 8; ++r) {
      float t = sacc[0][r];
      for (int nt = 1; nt < 8; ++nt) t = fmaxf(t, sacc[nt][r]);
      for (int o = 8; o > 0; o >>= 1) t = fmaxf(t, __shfl_xor(t, o, 32));
      float mn = fmaxf(mrow[r], t);
      float corr = __expf(mrow[r] - mn);
      mrow[r] = mn;
      float rs = 0.f;
      for (int nt = 0; nt < 8; ++nt) {
        float p = __expf(sacc[nt][r] - mn);
        sacc[nt][r] = p;
        rs += p;
      }
      for (int o = 8; o > 0; o >>= 1) rs += __shfl_xor(rs, o, 32);
      lrow[r] = lrow[r] * corr + rs;
      for (int j = 0; j < 5; ++j) oacc[j][r] *= corr;
    }

    // stage P (bf16) for the A side of P@V
    int prow = wid * 16 + ((lane >= 16) ? 8 : 0);
    for (int nt = 0; nt < 8; ++nt)
      for (int r = 0; r < 8; ++r)
        Ps[(prow + r) * 136 + nt * 16 + (lane & 15)] = f2bf(sacc[nt][r]);
    __syncthreads();

    // O += P @ V (K=128 in 4 steps, N=80 in 5 tiles)
    int arow2 = wid * 16 + (lane & 15);
    int pab = (lane < 16) ? 0 : 8;
    for (int kc = 0; kc < 4; ++kc) {
      Frag pa;
      const ush* pp = &Ps[arow2 * 136 + kc * 32 + pab];
      pa.u[0] = *(const uint4*)pp;
      pa.u[1] = *(const uint4*)(pp + 16);
      for (int j = 0; j < 5; ++j) {
        Frag vb;
        const ush* vp = &Vt[(j * 16 + nbase) * 136 + kc * 32 + kqb];
        vb.u[0] = *(const uint4*)vp;
        vb.u[1] = *(const uint4*)(vp + 16);
        oacc[j] = __builtin_amdgcn_wmma_f32_16x16x32_bf16(
            false, pa.v, false, vb.v, (short)0, oacc[j], false, false);
      }
    }
  }

  for (int j = 0; j < 5; ++j) {
    int col = j * 16 + (lane & 15);
    if (col < HD)
      for (int r = 0; r < 8; ++r) {
        size_t row = qrow0 + r + ((lane >= 16) ? 8 : 0);
        O[row * D + hc + col] = f2bf(oacc[j][r] / lrow[r]);
      }
  }
}

// ---------------------------------------------------------------- launch
extern "C" void kernel_launch(void* const* d_in, const int* in_sizes, int n_in,
                              void* d_out, int out_size, void* d_ws, size_t ws_size,
                              hipStream_t stream) {
  (void)in_sizes; (void)n_in; (void)out_size; (void)ws_size;
  const float* x     = (const float*)d_in[0];
  const float* c     = (const float*)d_in[1];
  const float* ln1_w = (const float*)d_in[2];
  const float* ln1_b = (const float*)d_in[3];
  const float* wq    = (const float*)d_in[4];
  const float* wk    = (const float*)d_in[5];
  const float* wv    = (const float*)d_in[6];
  const float* wo    = (const float*)d_in[7];
  const float* wo_b  = (const float*)d_in[8];
  const float* ln2_w = (const float*)d_in[9];
  const float* ln2_b = (const float*)d_in[10];
  const float* fc1_w = (const float*)d_in[11];
  const float* fc1_b = (const float*)d_in[12];
  const float* fc2_w = (const float*)d_in[13];
  const float* fc2_b = (const float*)d_in[14];
  const float* ada_w = (const float*)d_in[15];
  const float* ada_b = (const float*)d_in[16];

  const size_t DDn = (size_t)D * D;
  const size_t DHn = (size_t)D * HID;
  const size_t MDn = (size_t)M_ROWS * D;

  char* w = (char*)d_ws;
  ush* wqt = (ush*)w;  w += DDn * 2;       // all weights stored transposed [N][K]
  ush* wkt = (ush*)w;  w += DDn * 2;
  ush* wvt = (ush*)w;  w += DDn * 2;
  ush* wot = (ush*)w;  w += DDn * 2;
  ush* fc1t = (ush*)w; w += DHn * 2;
  ush* fc2t = (ush*)w; w += DHn * 2;
  float* modb = (float*)w; w += (size_t)BB * SIXD * 4;
  ush* h_bf = (ush*)w;   w += MDn * 2;     // reused for ln2 output
  ush* q_bf = (ush*)w;   w += MDn * 2;
  ush* k_bf = (ush*)w;   w += MDn * 2;
  ush* v_bf = (ush*)w;   w += MDn * 2;
  ush* attn_bf = (ush*)w; w += MDn * 2;
  float* proj   = (float*)q_bf;            // overlays q+k after attention
  ush*   mid    = q_bf;                    // overlays q,k,v,attn (= M*HID bf16)
  float* x1     = (float*)w; w += MDn * 4;
  float* mlpout = (float*)w; w += MDn * 4;

  // 1) weights -> bf16, transposed to [N][K]
  dim3 gtD(D / 32, D / 32);
  k_w_t_bf16<<<gtD, 256, 0, stream>>>(wq, wqt, D, D);
  k_w_t_bf16<<<gtD, 256, 0, stream>>>(wk, wkt, D, D);
  k_w_t_bf16<<<gtD, 256, 0, stream>>>(wv, wvt, D, D);
  k_w_t_bf16<<<gtD, 256, 0, stream>>>(wo, wot, D, D);
  k_w_t_bf16<<<dim3(HID / 32, D / 32), 256, 0, stream>>>(fc1_w, fc1t, D, HID);   // -> [HID][D]
  k_w_t_bf16<<<dim3(D / 32, HID / 32), 256, 0, stream>>>(fc2_w, fc2t, HID, D);   // -> [D][HID]

  // 2) adaLN modulation
  k_ada<<<dim3(SIXD / 256, BB), 256, 0, stream>>>(c, ada_w, ada_b, modb);

  // 3) LN1 + modulate -> h (bf16)
  k_ln_mod<<<M_ROWS, 256, 0, stream>>>(x, ln1_w, ln1_b, modb, 0, D, h_bf);

  // 4) q, k, v projections
  dim3 gD(D / 128, M_ROWS / 128);
  k_gemm<<<gD, 256, 0, stream>>>(h_bf, wqt, nullptr, q_bf, D, D, EP_BF16);
  k_gemm<<<gD, 256, 0, stream>>>(h_bf, wkt, nullptr, k_bf, D, D, EP_BF16);
  k_gemm<<<gD, 256, 0, stream>>>(h_bf, wvt, nullptr, v_bf, D, D, EP_BF16);

  // 5) attention (83.2 KB dynamic LDS; CDNA5 WGP has 320 KB)
  const int attn_smem = (128 * 104 + 80 * 136 + 128 * 136) * 2;
  (void)hipFuncSetAttribute((const void*)k_attn, hipFuncAttributeMaxDynamicSharedMemorySize, attn_smem);
  k_attn<<<BB * NH * (S / 128), 256, attn_smem, stream>>>(q_bf, k_bf, v_bf, attn_bf);

  // 6) output projection + gated residual
  k_gemm<<<gD, 256, 0, stream>>>(attn_bf, wot, wo_b, proj, D, D, EP_F32);
  k_residual<<<(int)(MDn / 256), 256, 0, stream>>>(x, proj, modb, 2 * D, x1);

  // 7) LN2 + modulate -> h2 (bf16, reuses h_bf)
  k_ln_mod<<<M_ROWS, 256, 0, stream>>>(x1, ln2_w, ln2_b, modb, 3 * D, 4 * D, h_bf);

  // 8) MLP
  dim3 gH(HID / 128, M_ROWS / 128);
  k_gemm<<<gH, 256, 0, stream>>>(h_bf, fc1t, fc1_b, mid, HID, D, EP_GELU_BF16);
  k_gemm<<<gD, 256, 0, stream>>>(mid, fc2t, fc2_b, mlpout, D, HID, EP_F32);
  k_residual<<<(int)(MDn / 256), 256, 0, stream>>>(x1, mlpout, modb, 5 * D, (float*)d_out);
}
